// MultiGranularityScorer_76836964925927
// MI455X (gfx1250) — matrix operand order repaired and checked by reading, not compiled, not run on hardware
//
#include <hip/hip_runtime.h>
#include <hip/hip_bf16.h>
#include <math.h>

// ---------------------------------------------------------------------------
// MultiGranularityScorer for MI455X (gfx1250, wave32, WMMA)
//
// d_in: 0 query_embeddings f32 (Nq,128)   1 doc_embeddings f32 (Nd,128)
//       2 query_mask bool(u8) (Nq)        3 doc_mask bool(u8) (Nd)
//       4 W2 f32 (128,128,2)  5 b2 f32(128)
//       6 W3 f32 (128,128,3)  7 b3 f32(128)
//       8 scale_logits f32 (3)
// d_out: f32 [0]=total, [1..Nq]=unigram_scores
//
// Note: explicit sched_barrier pipelining was tried and REJECTED — it forced
// both ping-pong buffers live past the allocator's occupancy-driven VGPR cap
// and caused scratch spills.  The plain double-buffered source below compiles
// spill-free; latency is hidden by the scheduler's one-tile-ahead load
// placement plus wave-level parallelism.
// ---------------------------------------------------------------------------

typedef __attribute__((ext_vector_type(16))) _Float16 v16h;
typedef __attribute__((ext_vector_type(8)))  _Float16 v8h;
typedef __attribute__((ext_vector_type(8)))  float    v8f;

#define DEVFN static __device__ __forceinline__

DEVFN int imin(int a, int b) { return a < b ? a : b; }

// Load a WMMA A-fragment (16x32 f16) for this lane from a row-major f16
// matrix. rowbase = pointer to this lane's row (already clamped), kb = K/32.
// Layout: lanes 0-15 hold K {0..15}; lanes 16-31 hold K {16..31} (matches B).
DEVFN v16h load_a_frag(const _Float16* rowbase, int kb, int half) {
  union { v16h v; v8h h2[2]; } u;
  u.h2[0] = *(const v8h*)(rowbase + kb * 32 + half * 16);
  u.h2[1] = *(const v8h*)(rowbase + kb * 32 + half * 16 + 8);
  return u.v;
}

// ---------------------------------------------------------------------------
// 1) f32 -> f16 elementwise convert (unigram embeddings double as n-gram A)
// ---------------------------------------------------------------------------
__global__ void k_cvt_f16(const float* __restrict__ src,
                          _Float16* __restrict__ dst, int n) {
  int i = blockIdx.x * blockDim.x + threadIdx.x;
  if (i < n) dst[i] = (_Float16)src[i];
}

// ---------------------------------------------------------------------------
// 2) Pack conv weights: WtT[o][j*128+d] = (f16) W[o][d][j]  (row-major NxK)
// ---------------------------------------------------------------------------
__global__ void k_pack_w(const float* __restrict__ W,
                         _Float16* __restrict__ WtT, int k) {
  int idx = blockIdx.x * blockDim.x + threadIdx.x;
  int kD = 128 * k;
  if (idx >= 128 * kD) return;
  int o = idx / kD, kk = idx % kD;
  int j = kk >> 7, d = kk & 127;
  WtT[idx] = (_Float16)W[(o * 128 + d) * k + j];
}

// ---------------------------------------------------------------------------
// 3) n-gram conv embed + L2 normalize.  out[i] = normalize(b + A_i @ WtT^T)
//    A_i = contiguous k*128 f16 strip starting at x16 + i*128.
//    One wave per 16-row tile; 8 col-slabs of 16; WMMA f16->f32 accumulate.
// ---------------------------------------------------------------------------
template <int K>
__global__ void k_ngram(const _Float16* __restrict__ x16,
                        const float* __restrict__ bias,
                        const _Float16* __restrict__ WtT,
                        _Float16* __restrict__ out16, int n) {
  constexpr int kD = 128 * K;
  constexpr int KB = kD / 32;  // 8 (k=2) or 12 (k=3), always even
  const int wave = threadIdx.x >> 5, lane = threadIdx.x & 31;
  const int half = lane >> 4, ln = lane & 15;
  const int tilebase = (blockIdx.x * 8 + wave) * 16;
  if (tilebase >= n) return;  // wave-uniform

  const int arow = imin(tilebase + ln, n - 1);
  const _Float16* abase = x16 + (size_t)arow * 128;

  v8f C[8];
#pragma unroll
  for (int s = 0; s < 8; s++) {
    float bv = bias[s * 16 + ln];
#pragma unroll
    for (int vg = 0; vg < 8; vg++) C[s][vg] = bv;
  }

  auto load_set = [&](int kb, v16h& a, v16h* b) {
    a = load_a_frag(abase, kb, half);
#pragma unroll
    for (int s = 0; s < 8; s++)
      b[s] = *(const v16h*)(WtT + (size_t)(s * 16 + ln) * kD + kb * 32 + half * 16);
  };
  auto compute = [&](const v16h& a, const v16h* b) {
#pragma unroll
    for (int s = 0; s < 8; s++)
      C[s] = __builtin_amdgcn_wmma_f32_16x16x32_f16(
          false, a, false, b[s], (short)0, C[s], false, false);
  };

  v16h a0, a1;
  v16h b0[8], b1[8];
  load_set(0, a0, b0);
#pragma unroll
  for (int kb = 0; kb < KB; kb += 2) {
    if (kb + 1 < KB) load_set(kb + 1, a1, b1);
    compute(a0, b0);
    if (kb + 2 < KB) load_set(kb + 2, a0, b0);
    if (kb + 1 < KB) compute(a1, b1);
  }

  // Per-row L2 norm: row (half*8+vg) lives on the 16-lane group across cols.
  float scale[8];
#pragma unroll
  for (int vg = 0; vg < 8; vg++) {
    float ss = 0.f;
#pragma unroll
    for (int s = 0; s < 8; s++) ss += C[s][vg] * C[s][vg];
    ss += __shfl_xor(ss, 1);
    ss += __shfl_xor(ss, 2);
    ss += __shfl_xor(ss, 4);
    ss += __shfl_xor(ss, 8);
    scale[vg] = 1.0f / fmaxf(sqrtf(ss), 1e-12f);
  }

#pragma unroll
  for (int vg = 0; vg < 8; vg++) {
    int r = tilebase + half * 8 + vg;
    if (r < n) {
#pragma unroll
      for (int s = 0; s < 8; s++)
        out16[(size_t)r * 128 + s * 16 + ln] = (_Float16)(C[s][vg] * scale[vg]);
    }
  }
}

// ---------------------------------------------------------------------------
// 4) Fused GEMM + masked row-max.
//    Block = 64 query rows (4 WMMA tiles held as A fragments per wave)
//            x one doc chunk (grid.y = NCHUNK).  8 waves stride doc tiles.
//    Each doc-tile B load (4 x 32B fragments per lane) feeds 16 WMMAs (4 qtiles x 4 K-steps),
//    and doc tiles are double-buffered (prefetch t+8 while computing t).
//    Writes partials[chunk * nqp + row] = max over chunk's valid doc cols.
// ---------------------------------------------------------------------------
template <int KNG>
__global__ void k_maxsim(const _Float16* __restrict__ q16,
                         const _Float16* __restrict__ d16,
                         const unsigned char* __restrict__ dmask,
                         int nq, int nd, int dpc,
                         float* __restrict__ partials, int nqp) {
  constexpr int QT = 4;  // query tiles per wave
  __shared__ float smax[8 * QT * 16];
  const int wave = threadIdx.x >> 5, lane = threadIdx.x & 31;
  const int half = lane >> 4, ln = lane & 15;
  const int qbase = blockIdx.x * (QT * 16);
  const float NEG_INF = -__builtin_huge_valf();

  // A fragments for 4 query tiles (rows qbase .. qbase+63), clamped.
  v16h a[QT][4];
#pragma unroll
  for (int qt = 0; qt < QT; qt++) {
    int qrow = imin(qbase + qt * 16 + ln, nq - 1);
    const _Float16* qb = q16 + (size_t)qrow * 128;
#pragma unroll
    for (int kb = 0; kb < 4; kb++) a[qt][kb] = load_a_frag(qb, kb, half);
  }

  const int dstart = blockIdx.y * dpc;
  const int dend = imin(nd, dstart + dpc);

  float rm[QT][8];
#pragma unroll
  for (int qt = 0; qt < QT; qt++)
#pragma unroll
    for (int vg = 0; vg < 8; vg++) rm[qt][vg] = NEG_INF;

  auto load_b = [&](int t, v16h* b) {
    int drow = imin(t * 16 + ln, nd - 1);
    const _Float16* db = d16 + (size_t)drow * 128;
#pragma unroll
    for (int kb = 0; kb < 4; kb++)
      b[kb] = *(const v16h*)(db + kb * 32 + half * 16);
  };
  auto compute = [&](int t, const v16h* b) {
    // Per-lane doc-column validity (after-WMMA mask; EXEC all-ones for WMMA).
    const int dcol = t * 16 + ln;
    bool valid = dcol < dend;
    if (valid) {
#pragma unroll
      for (int j = 0; j < KNG; j++) valid = valid && (dmask[dcol + j] != 0);
    }
#pragma unroll
    for (int qt = 0; qt < QT; qt++) {
      v8f C = {};
#pragma unroll
      for (int kb = 0; kb < 4; kb++)
        C = __builtin_amdgcn_wmma_f32_16x16x32_f16(
            false, a[qt][kb], false, b[kb], (short)0, C, false, false);
      if (valid) {
#pragma unroll
        for (int vg = 0; vg < 8; vg++) rm[qt][vg] = fmaxf(rm[qt][vg], C[vg]);
      }
    }
  };

  // Double-buffered sweep over this wave's doc tiles (wave-uniform control).
  v16h b0[4], b1[4];
  int dt = (dstart >> 4) + wave;
  if (dt * 16 < dend) load_b(dt, b0);
  for (; dt * 16 < dend; dt += 16) {
    if ((dt + 8) * 16 < dend) load_b(dt + 8, b1);
    compute(dt, b0);
    if ((dt + 16) * 16 < dend) load_b(dt + 16, b0);
    if ((dt + 8) * 16 < dend) compute(dt + 8, b1);
  }

  // Max across the 16 doc columns held by each 16-lane group, then waves.
#pragma unroll
  for (int qt = 0; qt < QT; qt++) {
#pragma unroll
    for (int vg = 0; vg < 8; vg++) {
      float v = rm[qt][vg];
      v = fmaxf(v, __shfl_xor(v, 1));
      v = fmaxf(v, __shfl_xor(v, 2));
      v = fmaxf(v, __shfl_xor(v, 4));
      v = fmaxf(v, __shfl_xor(v, 8));
      if (ln == 0) smax[(wave * QT + qt) * 16 + half * 8 + vg] = v;
    }
  }
  __syncthreads();
  if (threadIdx.x < QT * 16) {
    float m = smax[threadIdx.x];
#pragma unroll
    for (int w = 1; w < 8; w++) m = fmaxf(m, smax[w * QT * 16 + threadIdx.x]);
    int row = qbase + threadIdx.x;
    if (row < nqp) partials[(size_t)blockIdx.y * nqp + row] = m;
  }
}

// ---------------------------------------------------------------------------
// 5) Row-max over chunks, apply query mask, sum; optionally emit row scores.
// ---------------------------------------------------------------------------
__global__ void k_reduce(const float* __restrict__ partials, int nchunk,
                         int nqp, int nq, const unsigned char* __restrict__ qmask,
                         int kng, float* __restrict__ total_slot,
                         float* __restrict__ rowscores) {
  __shared__ float ssum[256];
  float local = 0.f;
  for (int row = threadIdx.x; row < nq; row += blockDim.x) {
    float m = -__builtin_huge_valf();
    for (int c = 0; c < nchunk; c++)
      m = fmaxf(m, partials[(size_t)c * nqp + row]);
    bool qv = true;
    for (int j = 0; j < kng; j++) qv = qv && (qmask[row + j] != 0);
    float s = qv ? m : 0.f;
    if (rowscores) rowscores[row] = s;
    local += s;
  }
  ssum[threadIdx.x] = local;
  __syncthreads();
  for (int off = 128; off > 0; off >>= 1) {
    if ((int)threadIdx.x < off) ssum[threadIdx.x] += ssum[threadIdx.x + off];
    __syncthreads();
  }
  if (threadIdx.x == 0) *total_slot = ssum[0];
}

// ---------------------------------------------------------------------------
// 6) softmax(scale_logits) . totals  -> d_out[0]
// ---------------------------------------------------------------------------
__global__ void k_combine(const float* __restrict__ logits,
                          const float* __restrict__ totals,
                          float* __restrict__ out0) {
  if (threadIdx.x == 0 && blockIdx.x == 0) {
    float a = logits[0], b = logits[1], c = logits[2];
    float mx = fmaxf(a, fmaxf(b, c));
    float e0 = expf(a - mx), e1 = expf(b - mx), e2 = expf(c - mx);
    float inv = 1.0f / (e0 + e1 + e2);
    out0[0] = inv * (e0 * totals[0] + e1 * totals[1] + e2 * totals[2]);
  }
}

// ---------------------------------------------------------------------------
extern "C" void kernel_launch(void* const* d_in, const int* in_sizes, int n_in,
                              void* d_out, int out_size, void* d_ws, size_t ws_size,
                              hipStream_t stream) {
  const float* qf = (const float*)d_in[0];
  const float* df = (const float*)d_in[1];
  const unsigned char* qm = (const unsigned char*)d_in[2];
  const unsigned char* dm = (const unsigned char*)d_in[3];
  const float* W2 = (const float*)d_in[4];
  const float* b2 = (const float*)d_in[5];
  const float* W3 = (const float*)d_in[6];
  const float* b3 = (const float*)d_in[7];
  const float* logits = (const float*)d_in[8];

  const int D = 128;
  const int Nq = in_sizes[0] / D;
  const int Nd = in_sizes[1] / D;
  float* out = (float*)d_out;

  // Bump allocator over workspace (256B aligned chunks). ~26 MB total.
  char* base = (char*)d_ws;
  size_t off = 0;
  auto alloc = [&](size_t bytes) -> void* {
    void* r = base + off;
    off = (off + bytes + 255) & ~(size_t)255;
    return r;
  };
  _Float16* q1 = (_Float16*)alloc((size_t)Nq * D * 2);
  _Float16* d1 = (_Float16*)alloc((size_t)Nd * D * 2);
  _Float16* q2 = (_Float16*)alloc((size_t)(Nq - 1) * D * 2);
  _Float16* d2 = (_Float16*)alloc((size_t)(Nd - 1) * D * 2);
  _Float16* q3 = (_Float16*)alloc((size_t)(Nq - 2) * D * 2);
  _Float16* d3 = (_Float16*)alloc((size_t)(Nd - 2) * D * 2);
  _Float16* Wt2 = (_Float16*)alloc((size_t)128 * 256 * 2);
  _Float16* Wt3 = (_Float16*)alloc((size_t)128 * 384 * 2);
  const int NCHUNK = 64;
  const int nqp = ((Nq + 15) / 16) * 16;
  float* part1 = (float*)alloc((size_t)NCHUNK * nqp * 4);
  float* part2 = (float*)alloc((size_t)NCHUNK * nqp * 4);
  float* part3 = (float*)alloc((size_t)NCHUNK * nqp * 4);
  float* totals = (float*)alloc(3 * 4);

  // 1) f32->f16 conversions
  k_cvt_f16<<<(Nq * D + 255) / 256, 256, 0, stream>>>(qf, q1, Nq * D);
  k_cvt_f16<<<(Nd * D + 255) / 256, 256, 0, stream>>>(df, d1, Nd * D);

  // 2) weight packs
  k_pack_w<<<(128 * 256 + 255) / 256, 256, 0, stream>>>(W2, Wt2, 2);
  k_pack_w<<<(128 * 384 + 255) / 256, 256, 0, stream>>>(W3, Wt3, 3);

  // 3) n-gram embeddings (WMMA) + L2 normalize
  auto ngrid = [](int n) { return (n + 16 * 8 - 1) / (16 * 8); };
  k_ngram<2><<<ngrid(Nq - 1), 256, 0, stream>>>(q1, b2, Wt2, q2, Nq - 1);
  k_ngram<2><<<ngrid(Nd - 1), 256, 0, stream>>>(d1, b2, Wt2, d2, Nd - 1);
  k_ngram<3><<<ngrid(Nq - 2), 256, 0, stream>>>(q1, b3, Wt3, q3, Nq - 2);
  k_ngram<3><<<ngrid(Nd - 2), 256, 0, stream>>>(d1, b3, Wt3, d3, Nd - 2);

  // 4) fused GEMM + rowmax per granularity (64 q-rows per block)
  auto dpc_of = [&](int nd) {
    int tiles = (nd + 15) / 16;
    return ((tiles + NCHUNK - 1) / NCHUNK) * 16;
  };
  {
    int nq = Nq, nd = Nd;
    dim3 g((nq + 63) / 64, NCHUNK);
    k_maxsim<1><<<g, 256, 0, stream>>>(q1, d1, dm, nq, nd, dpc_of(nd), part1, nqp);
  }
  {
    int nq = Nq - 1, nd = Nd - 1;
    dim3 g((nq + 63) / 64, NCHUNK);
    k_maxsim<2><<<g, 256, 0, stream>>>(q2, d2, dm, nq, nd, dpc_of(nd), part2, nqp);
  }
  {
    int nq = Nq - 2, nd = Nd - 2;
    dim3 g((nq + 63) / 64, NCHUNK);
    k_maxsim<3><<<g, 256, 0, stream>>>(q3, d3, dm, nq, nd, dpc_of(nd), part3, nqp);
  }

  // 5) reductions (unigram also emits per-row scores to d_out[1..Nq])
  k_reduce<<<1, 256, 0, stream>>>(part1, NCHUNK, nqp, Nq, qm, 1, totals + 0, out + 1);
  k_reduce<<<1, 256, 0, stream>>>(part2, NCHUNK, nqp, Nq - 1, qm, 2, totals + 1, nullptr);
  k_reduce<<<1, 256, 0, stream>>>(part3, NCHUNK, nqp, Nq - 2, qm, 3, totals + 2, nullptr);

  // 6) softmax combine -> d_out[0]
  k_combine<<<1, 1, 0, stream>>>(logits, totals, out);
}
